// DynamicFrequencyGNN_48335561949659
// MI455X (gfx1250) — matrix operand order repaired and checked by reading, not compiled, not run on hardware
//
#include <hip/hip_runtime.h>
#include <hip/hip_bf16.h>
#include <stdint.h>

// ---------------------------------------------------------------------------
// MI455X (gfx1250) DynamicFrequencyGNN.
// Roofline: P = 512 MB fp32 -> ~22us @ 23.3 TB/s, streamed from HBM exactly
// once by the fused P@[XW|X] kernel. The B panel is staged into LDS with
// double-buffered global_load_async_to_lds_b128 (ASYNCcnt) so panel DMA
// overlaps the WMMA burst. All GEMMs use v_wmma_f32_16x16x32_bf16 with
// fragment-major pre-packed B operands (one 32B vector load per fragment).
// ---------------------------------------------------------------------------

typedef __bf16 bf16_t;
typedef __attribute__((ext_vector_type(16))) __bf16 v16bf;
typedef __attribute__((ext_vector_type(8)))  float  v8f;

#define NT 8
#define NNODES 4096
#define FDIM 128
#define EEDGES 131072

__device__ __forceinline__ v8f vzero8() {
  v8f z;
#pragma unroll
  for (int i = 0; i < 8; ++i) z[i] = 0.f;
  return z;
}

__device__ __forceinline__ v8f wmma_bf16(v16bf a, v16bf b, v8f c) {
  return __builtin_amdgcn_wmma_f32_16x16x32_bf16(false, a, false, b, (short)0, c,
                                                 false, false);
}

// A fragment (16x32 bf16, row-major source). Two contiguous 8-elem runs/lane.
__device__ __forceinline__ v16bf load_a_bf16(const bf16_t* __restrict__ A, int lda) {
  int lane = threadIdx.x & 31;
  int row  = lane & 15;
  int kb   = (lane >> 4) << 3;
  const bf16_t* p = A + (size_t)row * lda + kb;
  v16bf a;
#pragma unroll
  for (int i = 0; i < 8; ++i) a[i] = p[i];
#pragma unroll
  for (int i = 0; i < 8; ++i) a[8 + i] = p[16 + i];
  return a;
}

// A fragment from fp32 source, converted to bf16 in-register.
__device__ __forceinline__ v16bf load_a_f32(const float* __restrict__ A, int lda) {
  int lane = threadIdx.x & 31;
  int row  = lane & 15;
  int kb   = (lane >> 4) << 3;
  const float* p = A + (size_t)row * lda + kb;
  v16bf a;
#pragma unroll
  for (int i = 0; i < 8; ++i) a[i] = (bf16_t)p[i];
#pragma unroll
  for (int i = 0; i < 8; ++i) a[8 + i] = (bf16_t)p[16 + i];
  return a;
}

// Packed-fragment B load: fragment base + lane*16 bf16, one 32B vector.
__device__ __forceinline__ v16bf load_b_pk(const bf16_t* __restrict__ Bpk, int frag) {
  int lane = threadIdx.x & 31;
  return *(const v16bf*)(Bpk + ((size_t)frag * 32 + lane) * 16);
}

__device__ __forceinline__ float sigmoidf_(float x) { return 1.f / (1.f + __expf(-x)); }

// ---------------------------------------------------------------------------
// Pack an f32 matrix into WMMA B-fragment-major bf16.
// trans=0: src row-major [K,N], ld=N.  trans=1: src row-major [N,K], ld=K (B=W^T).
__global__ void __launch_bounds__(256)
pack_b_kernel(const float* __restrict__ src, bf16_t* __restrict__ dst,
              int K, int N, int ld, int trans) {
  int i = blockIdx.x * blockDim.x + threadIdx.x;
  if (i >= K * N) return;
  int e     = i & 15;
  int lane  = (i >> 4) & 31;
  int f     = i >> 9;
  int ntile = N >> 4;
  int nt = f % ntile, kt = f / ntile;
  int col = lane & 15, khalf = lane >> 4;
  int k = kt * 32 + (e < 8 ? khalf * 8 + e : 16 + khalf * 8 + (e - 8));
  int n = nt * 16 + col;
  float v = trans ? src[(size_t)n * ld + k] : src[(size_t)k * ld + n];
  dst[i] = (bf16_t)v;
}

// Pack features (fp32 [T*N,128]) into panel fragments nt 8..15.
// Panel layout: panel[((t*128 + kt)*16 + nt)*512 + lane*16 + e].
__global__ void __launch_bounds__(256)
pack_panel_x_kernel(const float* __restrict__ X, bf16_t* __restrict__ panel) {
  int i = blockIdx.x * blockDim.x + threadIdx.x;  // < T*4096*128
  int e    = i & 15;
  int lane = (i >> 4) & 31;
  int fidx = i >> 9;            // t*1024 + kt*8 + j
  int j    = fidx & 7;
  int kt   = (fidx >> 3) & 127;
  int t    = fidx >> 10;
  int col = lane & 15, khalf = lane >> 4;
  int k  = kt * 32 + (e < 8 ? khalf * 8 + e : 16 + khalf * 8 + (e - 8));
  int fd = j * 16 + col;
  float v = X[((size_t)t * NNODES + k) * FDIM + fd];
  size_t d = ((((size_t)t * 128 + kt) * 16 + (8 + j)) * 32 + lane) * 16 + e;
  panel[d] = (bf16_t)v;
}

// ---------------------------------------------------------------------------
// GEMM with N=128: acc = A[M,K] @ B[K,128], B pre-packed fragment-major.
// AF32   : A is fp32, converted in-register.
// PACKOUT: store packed into panel fragments nt 0..7 (XW path, no bias);
//          else row-major bf16 with bias+relu (high-pass path).
template <bool AF32, bool PACKOUT>
__global__ void __launch_bounds__(256)
gemm_n128_kernel(const void* __restrict__ Aptr, const bf16_t* __restrict__ Bpk,
                 const float* __restrict__ bias, bf16_t* __restrict__ outRM,
                 int ldo, bf16_t* __restrict__ outPK, int M, int K) {
  int wave = (int)((blockIdx.x * blockDim.x + threadIdx.x) >> 5);
  int m0 = wave * 16;
  if (m0 >= M) return;
  v8f acc[8];
#pragma unroll
  for (int j = 0; j < 8; ++j) acc[j] = vzero8();

  int nkt = K >> 5;
  for (int kt = 0; kt < nkt; ++kt) {
    int k0 = kt << 5;
    v16bf a;
    if constexpr (AF32)
      a = load_a_f32((const float*)Aptr + (size_t)m0 * K + k0, K);
    else
      a = load_a_bf16((const bf16_t*)Aptr + (size_t)m0 * K + k0, K);
    v16bf bfr[8];
#pragma unroll
    for (int j = 0; j < 8; ++j) bfr[j] = load_b_pk(Bpk, kt * 8 + j);
#pragma unroll
    for (int j = 0; j < 8; ++j) acc[j] = wmma_bf16(a, bfr[j], acc[j]);
  }

  int lane = threadIdx.x & 31;
  int col  = lane & 15;
  int rofs = (lane >> 4) << 3;
  if constexpr (PACKOUT) {
    int t     = m0 >> 12;
    int kloc  = m0 & (NNODES - 1);
    int ktO   = kloc >> 5;
    int khalf = (kloc >> 4) & 1;
#pragma unroll
    for (int j = 0; j < 8; ++j) {
      bf16_t tmp[8];
#pragma unroll
      for (int r = 0; r < 8; ++r) tmp[r] = (bf16_t)acc[j][r];
      bf16_t* dst = outPK +
          ((((size_t)t * 128 + ktO) * 16 + j) * 32 + (rofs >> 3) * 16 + col) * 16 + khalf * 8;
      *(uint4*)dst = *(const uint4*)tmp;
    }
  } else {
#pragma unroll
    for (int j = 0; j < 8; ++j) {
      float bv = bias[j * 16 + col];
#pragma unroll
      for (int r = 0; r < 8; ++r) {
        float v = fmaxf(acc[j][r] + bv, 0.f);
        outRM[(size_t)(m0 + rofs + r) * ldo + j * 16 + col] = (bf16_t)v;
      }
    }
  }
}

// ---------------------------------------------------------------------------
// Fused frequency filter: per t, 128-row block computes P@[XW|X].
// Panel (fragment-major in global) is staged into LDS with double-buffered
// global_load_async_to_lds_b128; next K-panel DMA overlaps the WMMA burst.
__global__ void __launch_bounds__(256)
freq_filter_kernel(const float* __restrict__ P, const float* __restrict__ X,
                   const bf16_t* __restrict__ panel, const float* __restrict__ b_lpf,
                   bf16_t* __restrict__ xseq, bf16_t* __restrict__ Dbf) {
  __shared__ __align__(32) bf16_t Bsh[2][8192];  // 2 x 16 KB double buffer
  int t    = blockIdx.y;
  int w    = threadIdx.x >> 5;
  int lane = threadIdx.x & 31;
  int m0   = blockIdx.x * 128 + w * 16;

  v8f acc[16];
#pragma unroll
  for (int j = 0; j < 16; ++j) acc[j] = vzero8();

  const float* Pt = P + (size_t)t * NNODES * NNODES;
  size_t rowbase  = (size_t)t * NNODES;
  int arow = lane & 15;
  int kba  = (lane >> 4) << 3;
  const bf16_t* panT = panel + (size_t)t * 128 * 8192;

  auto stage = [&](int kt, int buf) {
    const bf16_t* psrc = panT + (size_t)kt * 8192;
#pragma unroll
    for (int rep = 0; rep < 4; ++rep) {
      int idx = (rep * 256 + (int)threadIdx.x) * 8;   // 16B per op, 16 KB total
      unsigned dst = (unsigned)(unsigned long long)(uintptr_t)(&Bsh[buf][idx]);
      asm volatile("global_load_async_to_lds_b128 %0, %1, off"
                   :: "v"(dst), "v"(psrc + idx) : "memory");
    }
  };

  stage(0, 0);
  asm volatile("s_wait_asynccnt 0" ::: "memory");
  __syncthreads();

  for (int kt = 0; kt < 128; ++kt) {
    int cur = kt & 1;
    if (kt + 1 < 128) stage(kt + 1, cur ^ 1);   // overlap next panel DMA

    const float* pa = Pt + (size_t)(m0 + arow) * NNODES + (kt * 32 + kba);
    __builtin_prefetch(pa + 32, 0, 0);          // P stream (HBM roofline term)
    v16bf a;
#pragma unroll
    for (int i = 0; i < 8; ++i) a[i] = (bf16_t)pa[i];
#pragma unroll
    for (int i = 0; i < 8; ++i) a[8 + i] = (bf16_t)pa[16 + i];

#pragma unroll
    for (int g = 0; g < 2; ++g) {
      v16bf bfr[8];
#pragma unroll
      for (int j = 0; j < 8; ++j)
        bfr[j] = *(const v16bf*)(&Bsh[cur][(g * 8 + j) * 512 + lane * 16]);
#pragma unroll
      for (int j = 0; j < 8; ++j)
        acc[g * 8 + j] = wmma_bf16(a, bfr[j], acc[g * 8 + j]);
    }
    asm volatile("s_wait_asynccnt 0" ::: "memory");
    __syncthreads();
  }

  int col  = lane & 15;
  int rofs = (lane >> 4) << 3;
  // low = relu(P@XW + b_lpf) -> xseq[:, 0:128]
#pragma unroll
  for (int j = 0; j < 8; ++j) {
    float bv = b_lpf[j * 16 + col];
#pragma unroll
    for (int r = 0; r < 8; ++r) {
      float v = fmaxf(acc[j][r] + bv, 0.f);
      xseq[(rowbase + m0 + rofs + r) * 256 + j * 16 + col] = (bf16_t)v;
    }
  }
  // diff = X - P@X -> Dbf
#pragma unroll
  for (int j = 8; j < 16; ++j) {
    int fd = (j - 8) * 16 + col;
#pragma unroll
    for (int r = 0; r < 8; ++r) {
      size_t ridx = rowbase + m0 + rofs + r;
      float d = X[ridx * FDIM + fd] - acc[j][r];
      Dbf[ridx * FDIM + fd] = (bf16_t)d;
    }
  }
}

// ---------------------------------------------------------------------------
// One LSTM step: wave owns 16 rows x one gate-col tile j0 across all 4 gates.
__global__ void __launch_bounds__(256)
lstm_step_kernel(const bf16_t* __restrict__ xseq, int t,
                 const bf16_t* __restrict__ h_in, bf16_t* __restrict__ h_out,
                 float* __restrict__ c_ws,
                 const bf16_t* __restrict__ WihPk, const bf16_t* __restrict__ WhhPk,
                 const float* __restrict__ b_ih, const float* __restrict__ b_hh) {
  int wave = (int)((blockIdx.x * blockDim.x + threadIdx.x) >> 5);
  int mt = wave >> 4;
  int jt = wave & 15;
  int m0 = mt * 16, j0 = jt * 16;

  v8f acc[4];
#pragma unroll
  for (int g = 0; g < 4; ++g) acc[g] = vzero8();

  const bf16_t* x = xseq + ((size_t)t * NNODES + m0) * 256;
  for (int kt = 0; kt < 8; ++kt) {
    v16bf a = load_a_bf16(x + kt * 32, 256);
    v16bf bfr[4];
#pragma unroll
    for (int g = 0; g < 4; ++g) bfr[g] = load_b_pk(WihPk, kt * 64 + g * 16 + jt);
#pragma unroll
    for (int g = 0; g < 4; ++g) acc[g] = wmma_bf16(a, bfr[g], acc[g]);
  }
  if (t > 0) {
    const bf16_t* hp = h_in + (size_t)m0 * 256;
    for (int kt = 0; kt < 8; ++kt) {
      v16bf a = load_a_bf16(hp + kt * 32, 256);
      v16bf bfr[4];
#pragma unroll
      for (int g = 0; g < 4; ++g) bfr[g] = load_b_pk(WhhPk, kt * 64 + g * 16 + jt);
#pragma unroll
      for (int g = 0; g < 4; ++g) acc[g] = wmma_bf16(a, bfr[g], acc[g]);
    }
  }

  int lane = threadIdx.x & 31;
  int col  = lane & 15;
  int rofs = (lane >> 4) << 3;
  int n    = j0 + col;
  float bi = b_ih[n] + b_hh[n];
  float bf = b_ih[256 + n] + b_hh[256 + n];
  float bg = b_ih[512 + n] + b_hh[512 + n];
  float bo = b_ih[768 + n] + b_hh[768 + n];
#pragma unroll
  for (int r = 0; r < 8; ++r) {
    size_t m = (size_t)(m0 + rofs + r);
    float iv = sigmoidf_(acc[0][r] + bi);
    float fv = sigmoidf_(acc[1][r] + bf);
    float gv = tanhf(acc[2][r] + bg);
    float ov = sigmoidf_(acc[3][r] + bo);
    float cold = (t == 0) ? 0.f : c_ws[m * 256 + n];
    float c = fv * cold + iv * gv;
    c_ws[m * 256 + n] = c;
    h_out[m * 256 + n] = (bf16_t)(ov * tanhf(c));
  }
}

// ---------------------------------------------------------------------------
// Link predictor, fully fused: gathered A fragments, 16-tile WMMA GEMM vs
// packed W1, relu+bias, then 256->1 dot via in-wave shuffle reduction.
__global__ void __launch_bounds__(256)
linkpred_kernel(const bf16_t* __restrict__ h, const long long* __restrict__ eidx,
                const bf16_t* __restrict__ W1Pk, const float* __restrict__ b1,
                const float* __restrict__ W2, const float* __restrict__ b2,
                float* __restrict__ out) {
  int wave = (int)((blockIdx.x * blockDim.x + threadIdx.x) >> 5);
  int e0 = wave * 16;
  if (e0 >= EEDGES) return;

  int lane = threadIdx.x & 31;
  int arow = lane & 15;
  int kb   = (lane >> 4) << 3;
  int srci = (int)eidx[e0 + arow];
  int dsti = (int)eidx[EEDGES + e0 + arow];

  v8f acc[16];
#pragma unroll
  for (int j = 0; j < 16; ++j) acc[j] = vzero8();

  for (int kt = 0; kt < 16; ++kt) {
    int k0 = kt * 32;
    const bf16_t* base = (k0 < 256) ? (h + (size_t)srci * 256 + k0)
                                    : (h + (size_t)dsti * 256 + (k0 - 256));
    const bf16_t* pa = base + kb;
    v16bf a;
#pragma unroll
    for (int i = 0; i < 8; ++i) a[i] = pa[i];
#pragma unroll
    for (int i = 0; i < 8; ++i) a[8 + i] = pa[16 + i];
#pragma unroll
    for (int g = 0; g < 2; ++g) {
      v16bf bfr[8];
#pragma unroll
      for (int j = 0; j < 8; ++j) bfr[j] = load_b_pk(W1Pk, kt * 16 + g * 8 + j);
#pragma unroll
      for (int j = 0; j < 8; ++j)
        acc[g * 8 + j] = wmma_bf16(a, bfr[j], acc[g * 8 + j]);
    }
  }

  int col  = lane & 15;
  int rofs = (lane >> 4) << 3;
  float part[8];
#pragma unroll
  for (int r = 0; r < 8; ++r) part[r] = 0.f;
#pragma unroll
  for (int j = 0; j < 16; ++j) {
    int n = j * 16 + col;
    float w2 = W2[n];
    float bb = b1[n];
#pragma unroll
    for (int r = 0; r < 8; ++r)
      part[r] += fmaxf(acc[j][r] + bb, 0.f) * w2;
  }
#pragma unroll
  for (int off = 1; off < 16; off <<= 1) {
#pragma unroll
    for (int r = 0; r < 8; ++r) part[r] += __shfl_xor(part[r], off, 32);
  }
  if (col == 0) {
    float b2v = b2[0];
#pragma unroll
    for (int r = 0; r < 8; ++r) out[e0 + rofs + r] = part[r] + b2v;
  }
}

// ---------------------------------------------------------------------------
extern "C" void kernel_launch(void* const* d_in, const int* in_sizes, int n_in,
                              void* d_out, int out_size, void* d_ws, size_t ws_size,
                              hipStream_t stream) {
  (void)in_sizes; (void)n_in; (void)out_size; (void)ws_size;
  const float*     features = (const float*)d_in[0];
  const float*     P        = (const float*)d_in[1];
  const long long* eidx     = (const long long*)d_in[2];
  const float*     W_lpf    = (const float*)d_in[3];
  const float*     b_lpf    = (const float*)d_in[4];
  const float*     W_hpf    = (const float*)d_in[5];
  const float*     b_hpf    = (const float*)d_in[6];
  const float*     W_ih     = (const float*)d_in[7];
  const float*     W_hh     = (const float*)d_in[8];
  const float*     b_ih     = (const float*)d_in[9];
  const float*     b_hh     = (const float*)d_in[10];
  const float*     W1       = (const float*)d_in[11];
  const float*     b1       = (const float*)d_in[12];
  const float*     W2       = (const float*)d_in[13];
  const float*     b2       = (const float*)d_in[14];
  float*           out      = (float*)d_out;

  // workspace (~49 MB)
  bf16_t* panel   = (bf16_t*)d_ws;           // [T,128kt,16nt,32,16] 16 MB
  bf16_t* Dbf     = panel   + 8388608;       // [T*N,128]  8 MB
  bf16_t* xseq    = Dbf     + 4194304;       // [T*N,256] 16 MB
  bf16_t* Wlpf_pk = xseq    + 8388608;       // 32 KB
  bf16_t* Whpf_pk = Wlpf_pk + 16384;         // 32 KB
  bf16_t* Wih_pk  = Whpf_pk + 16384;         // 512 KB
  bf16_t* Whh_pk  = Wih_pk  + 262144;        // 512 KB
  bf16_t* W1_pk   = Whh_pk  + 262144;        // 256 KB
  bf16_t* hA      = W1_pk   + 131072;        // [N,256] 2 MB
  bf16_t* hB      = hA      + 1048576;       // [N,256] 2 MB
  float*  c_ws    = (float*)(hB + 1048576);  // [N,256] 4 MB

  // 1) pack weights into B-fragment-major bf16 (one-time)
  pack_b_kernel<<<64,   256, 0, stream>>>(W_lpf, Wlpf_pk, 128, 128, 128, 0);
  pack_b_kernel<<<64,   256, 0, stream>>>(W_hpf, Whpf_pk, 128, 128, 128, 0);
  pack_b_kernel<<<1024, 256, 0, stream>>>(W_ih, Wih_pk, 256, 1024, 256, 1);
  pack_b_kernel<<<1024, 256, 0, stream>>>(W_hh, Whh_pk, 256, 1024, 256, 1);
  pack_b_kernel<<<512,  256, 0, stream>>>(W1, W1_pk, 512, 256, 256, 0);

  // 2) pack X into panel fragments nt 8..15
  pack_panel_x_kernel<<<16384, 256, 0, stream>>>(features, panel);

  // 3) XW = X @ W_lpf (fp32 A converted in-register), packed into panel nt 0..7
  gemm_n128_kernel<true, true><<<256, 256, 0, stream>>>(
      features, Wlpf_pk, nullptr, nullptr, 0, panel, NT * NNODES, FDIM);

  // 4) fused P@[XW|X]: low -> xseq[:, :128], diff = X - PX -> Dbf
  freq_filter_kernel<<<dim3(NNODES / 128, NT), 256, 0, stream>>>(
      P, features, panel, b_lpf, xseq, Dbf);

  // 5) high = relu(diff @ W_hpf + b_hpf) -> xseq[:, 128:256]
  gemm_n128_kernel<false, false><<<256, 256, 0, stream>>>(
      Dbf, Whpf_pk, b_hpf, xseq + 128, 256, nullptr, NT * NNODES, FDIM);

  // 6) LSTM over T=8 steps (double-buffered h)
  bf16_t* hbuf[2] = {hA, hB};
  for (int t = 0; t < NT; ++t) {
    lstm_step_kernel<<<512, 256, 0, stream>>>(xseq, t, hbuf[t & 1],
                                              hbuf[(t + 1) & 1], c_ws,
                                              Wih_pk, Whh_pk, b_ih, b_hh);
  }
  bf16_t* hfin = hbuf[0];

  // 7) fused link predictor
  linkpred_kernel<<<1024, 256, 0, stream>>>(hfin, eidx, W1_pk, b1, W2, b2, out);
}
